// ScaledDotProductAttention_32856499815116
// MI455X (gfx1250) — matrix operand order repaired and checked by reading, not compiled, not run on hardware
//
#include <hip/hip_runtime.h>

typedef _Float16 v16h __attribute__((ext_vector_type(16)));
typedef _Float16 v8h  __attribute__((ext_vector_type(8)));
typedef __fp16   v2hf __attribute__((ext_vector_type(2)));
typedef float    v8f  __attribute__((ext_vector_type(8)));

#define BATCH 32
#define SEQ   2048
#define DIM   64
#define BQ    128      // query rows per block (8 waves x 16)
#define BK    128      // keys per step
#define NW    8        // waves per block

#define KSTR  72       // Klds row stride in halves (144B, 16B aligned, conflict-free)
#define VSTR  136      // Vt  row stride in halves (272B, 16B aligned, conflict-free)
#define PSTR  136      // P   row stride in halves

#define WMMA16(A, B, C) __builtin_amdgcn_wmma_f32_16x16x32_f16(false, A, false, B, (short)0, C, false, false)

// max-reduce over each 16-lane half using ds_swizzle xor patterns (CDNA5 §12.12)
__device__ __forceinline__ float swz_max16(float v) {
    v = fmaxf(v, __int_as_float(__builtin_amdgcn_ds_swizzle(__float_as_int(v), 0x041F))); // xor 1
    v = fmaxf(v, __int_as_float(__builtin_amdgcn_ds_swizzle(__float_as_int(v), 0x081F))); // xor 2
    v = fmaxf(v, __int_as_float(__builtin_amdgcn_ds_swizzle(__float_as_int(v), 0x101F))); // xor 4
    v = fmaxf(v, __int_as_float(__builtin_amdgcn_ds_swizzle(__float_as_int(v), 0x201F))); // xor 8
    return v;
}

__global__ __launch_bounds__(256)
void fa_fwd_kernel(const float* __restrict__ Q, const float* __restrict__ K,
                   const float* __restrict__ V, float* __restrict__ Out)
{
    __shared__ _Float16 Klds[BK * KSTR];        // [key][d]  row-major f16
    __shared__ _Float16 Vt[DIM * VSTR];         // [d][key]  transposed f16
    __shared__ _Float16 Plds[NW * 16 * PSTR];   // per-wave prob scratch (16 x 128)

    const int tid  = threadIdx.x;
    const int wave = tid >> 5;
    const int lane = tid & 31;
    const int m    = lane & 15;   // row / column-within-tile index
    const int h    = lane >> 4;   // lane-half
    const int b    = blockIdx.y;
    const int qrow = blockIdx.x * BQ + wave * 16;

    // 1/sqrt(64) * log2(e): softmax in exp2 domain, scale folded into Q
    const float scale = 0.125f * 1.44269504088896340736f;

    // ---- Q tile (16x64) -> two f16 A-operands (16x32) ----
    // A layout: lane-half h: VGPR j(0..7) K=16h+2j
    v16h qa[2];
    {
        const float* qb = Q + ((size_t)b * SEQ + (qrow + m)) * DIM;
        #pragma unroll
        for (int c = 0; c < 2; ++c) {
            #pragma unroll
            for (int jj = 0; jj < 8; ++jj) {
                int k = 32 * c + 16 * h + 2 * jj;
                float2 xy = *(const float2*)(qb + k);
                v2hf pk = __builtin_amdgcn_cvt_pkrtz(xy.x * scale, xy.y * scale);
                qa[c][2 * jj]     = (_Float16)pk[0];
                qa[c][2 * jj + 1] = (_Float16)pk[1];
            }
        }
    }

    // all-ones B operand: row-sum of P via WMMA (every output column = row sum)
    v16h onesb;
    #pragma unroll
    for (int i = 0; i < 16; ++i) onesb[i] = (_Float16)1.0f;

    v8f o[4];
    #pragma unroll
    for (int nt = 0; nt < 4; ++nt) o[nt] = (v8f){};
    float mrow[8], lrow[8];
    #pragma unroll
    for (int r = 0; r < 8; ++r) { mrow[r] = -3.0e38f; lrow[r] = 0.0f; }

    _Float16* myP = &Plds[wave * 16 * PSTR];

    const float* Kb = K + (size_t)b * SEQ * DIM;
    const float* Vb = V + (size_t)b * SEQ * DIM;

    for (int kt = 0; kt < SEQ / BK; ++kt) {
        // ---- K staging: 128 keys x 64 d, fp32 -> f16, row-major ----
        {
            int key   = tid >> 1;             // 0..127
            int dhalf = (tid & 1) * 32;       // 0 or 32
            const float* kp = Kb + (size_t)(kt * BK + key) * DIM + dhalf;
            #pragma unroll
            for (int i = 0; i < 8; ++i) {
                float4 kq = *(const float4*)(kp + 4 * i);
                *(v2hf*)&Klds[key * KSTR + dhalf + 4 * i]     = __builtin_amdgcn_cvt_pkrtz(kq.x, kq.y);
                *(v2hf*)&Klds[key * KSTR + dhalf + 4 * i + 2] = __builtin_amdgcn_cvt_pkrtz(kq.z, kq.w);
            }
            if (kt + 1 < SEQ / BK) __builtin_prefetch(kp + BK * DIM, 0, 1);
        }
        // ---- V staging: key-pairs packed with cvt_pk, stored transposed [d][key] ----
        {
            int d = (lane & 31) * 2;          // covers d, d+1 (0..63)
            #pragma unroll
            for (int i = 0; i < 8; ++i) {
                int k0 = 2 * (wave + 8 * i);  // key pairs 0..127
                const float* vp0 = Vb + (size_t)(kt * BK + k0) * DIM + d;
                float2 a0 = *(const float2*)(vp0);
                float2 a1 = *(const float2*)(vp0 + DIM);
                *(v2hf*)&Vt[(d + 0) * VSTR + k0] = __builtin_amdgcn_cvt_pkrtz(a0.x, a1.x);
                *(v2hf*)&Vt[(d + 1) * VSTR + k0] = __builtin_amdgcn_cvt_pkrtz(a0.y, a1.y);
            }
            if (kt + 1 < SEQ / BK)
                __builtin_prefetch(Vb + (size_t)(kt * BK + BK + 2 * wave) * DIM + d, 0, 1);
        }
        __syncthreads();

        // ---- Scores S = Q*K^T : eight 16x16 C tiles over 128 keys ----
        v8f c[8];
        #pragma unroll
        for (int t = 0; t < 8; ++t) c[t] = (v8f){};
        #pragma unroll
        for (int t = 0; t < 8; ++t) {
            #pragma unroll
            for (int cc = 0; cc < 2; ++cc) {
                // B layout (32x16): lane = column (key), lane-half h holds K(=d) 16h..16h+15
                v16h kb;
                const _Float16* kr = &Klds[(16 * t + m) * KSTR + 32 * cc + 16 * h];
                *(v8h*)&kb       = *(const v8h*)kr;
                *((v8h*)&kb + 1) = *(const v8h*)(kr + 8);
                c[t] = WMMA16(qa[cc], kb, c[t]);
            }
        }

        // ---- Online softmax. C layout: VGPR r -> row M=r+8h, lane%16 -> key ----
        #pragma unroll
        for (int r = 0; r < 8; ++r) {
            float v0 = fmaxf(fmaxf(c[0][r], c[1][r]), fmaxf(c[2][r], c[3][r]));
            float v1 = fmaxf(fmaxf(c[4][r], c[5][r]), fmaxf(c[6][r], c[7][r]));
            float v = swz_max16(fmaxf(v0, v1));
            float mn   = fmaxf(mrow[r], v);
            float corr = exp2f(mrow[r] - mn);
            mrow[r] = mn;
            lrow[r] *= corr;
            o[0][r] *= corr; o[1][r] *= corr; o[2][r] *= corr; o[3][r] *= corr;
            int M = r + 8 * h;
            #pragma unroll
            for (int t = 0; t < 8; ++t) {
                float p = exp2f(c[t][r] - mn);
                myP[M * PSTR + 16 * t + m] = (_Float16)p;   // C layout -> LDS
            }
        }
        asm volatile("s_wait_dscnt 0" ::: "memory");

        // ---- Per 32-key chunk: reload P in A layout, row-sum via ones-WMMA, O += P*V ----
        v8f lt = (v8f){};
        #pragma unroll
        for (int t2 = 0; t2 < 4; ++t2) {
            v16h pa;
            {
                const _Float16* pr = &myP[m * PSTR + 32 * t2];
                *(v8h*)&pa       = *(const v8h*)(pr + 16 * h);
                *((v8h*)&pa + 1) = *(const v8h*)(pr + 16 * h + 8);
            }
            lt = WMMA16(pa, onesb, lt);
            #pragma unroll
            for (int nt = 0; nt < 4; ++nt) {
                v16h vb;  // B: lane = column d=16nt+m, lane-half h holds keys 32t2+16h..+15
                const _Float16* vr = &Vt[(16 * nt + m) * VSTR + 32 * t2 + 16 * h];
                *(v8h*)&vb       = *(const v8h*)vr;
                *((v8h*)&vb + 1) = *(const v8h*)(vr + 8);
                o[nt] = WMMA16(pa, vb, o[nt]);
            }
        }
        #pragma unroll
        for (int r = 0; r < 8; ++r) lrow[r] += lt[r];
        __syncthreads();
    }

    // ---- Epilogue: normalize, coalesced fp32 stores ----
    #pragma unroll
    for (int r = 0; r < 8; ++r) {
        float inv = 1.0f / lrow[r];
        int row = qrow + r + 8 * h;
        float* ob = Out + ((size_t)b * SEQ + row) * DIM + m;
        ob[0]  = o[0][r] * inv;
        ob[16] = o[1][r] * inv;
        ob[32] = o[2][r] * inv;
        ob[48] = o[3][r] * inv;
    }
}

extern "C" void kernel_launch(void* const* d_in, const int* in_sizes, int n_in,
                              void* d_out, int out_size, void* d_ws, size_t ws_size,
                              hipStream_t stream) {
    (void)in_sizes; (void)n_in; (void)out_size; (void)d_ws; (void)ws_size;
    const float* Q = (const float*)d_in[0];
    const float* K = (const float*)d_in[1];
    const float* V = (const float*)d_in[2];
    float* Out = (float*)d_out;
    dim3 grid(SEQ / BQ, BATCH);
    fa_fwd_kernel<<<grid, 256, 0, stream>>>(Q, K, V, Out);
}
